// TokenDistanceModule_11811160064732
// MI455X (gfx1250) — compile-verified, matrix operand order checked by reading, not run on hardware
//
#include <hip/hip_runtime.h>
#include <hip/hip_bf16.h>
#include <cstdint>

// MI455X fused TokenDistance kernel.
// Memory-bound: ~1.03 GB streaming traffic -> ~44us floor @ 23.3 TB/s;
// 34.4 GFLOP in f16 WMMA (f32 accum) is negligible on CDNA5 matrix units.
// z is read once / u written once -> non-temporal hints keep the small
// weights hot in L2; each block processes 256 rows so the per-block weight
// f32->f16 staging (64 KB from L2) is amortized over 128 KB of z stream.

typedef _Float16 v16h __attribute__((ext_vector_type(16)));
typedef _Float16 v8h  __attribute__((ext_vector_type(8)));
typedef float    v8f  __attribute__((ext_vector_type(8)));
typedef float    v4f  __attribute__((ext_vector_type(4)));

#define NTOK 1024
#define TZc  128      // pair channel dim (z / u)
#define TDc  64       // hidden dim
#define NBINS 38
#define WAVES 4
#define ROWS_PER_WAVE 16
#define TILES 4       // row-tiles per wave (amortizes weight staging)
#define ROWS_PER_BLOCK (WAVES * ROWS_PER_WAVE * TILES)

// Load a 16x32 f16 WMMA A/B fragment from an LDS row-major matrix.
// ISA 16-bit A layout (CDNA5 ISA): lanes 0-15 hold K=[k0..k0+7] in halves 0-7
// and K=[k0+16..k0+23] in halves 8-15; lanes 16-31 hold K offset by +8.
// Each 8-half run is 16B contiguous -> two ds_load_b128 per fragment.
__device__ __forceinline__ v16h lds_frag(const _Float16* row0, int stride,
                                         int k0, int lane) {
  const int hi = (lane >> 4) & 1;
  const _Float16* p = row0 + (lane & 15) * stride + k0 + hi * 8;
  union { v16h v; v8h h[2]; } f;
  f.h[0] = *(const v8h*)(p);
  f.h[1] = *(const v8h*)(p + 16);
  return f.v;
}

__global__ __launch_bounds__(128) void token_distance_fused(
    const float* __restrict__ z, const float* __restrict__ coords,
    const float* __restrict__ tdm, const float* __restrict__ Wz,
    const float* __restrict__ Wa, const float* __restrict__ Wu,
    const float* __restrict__ zg, const float* __restrict__ zb,
    const float* __restrict__ vg, const float* __restrict__ vb,
    float* __restrict__ out)
{
  __shared__ _Float16 sWz[TDc * TZc];                       // 16 KB  (B1: B[k=c,n=d] = Wz[d][c])
  __shared__ _Float16 sWu[TZc * TDc];                       // 16 KB  (B2: B[k=d,n=c] = Wu[c][d])
  __shared__ _Float16 sZn[WAVES * ROWS_PER_WAVE * TZc];     // 16 KB  normalized z rows (A, GEMM1)
  __shared__ _Float16 sRv[WAVES * ROWS_PER_WAVE * TDc];     //  8 KB  relu(LN(2v))     (A, GEMM2)
  __shared__ float sZg[TZc], sZb[TZc], sVg[TDc], sVb[TDc];
  __shared__ int   sIdx[WAVES * ROWS_PER_WAVE];
  __shared__ float sMsk[WAVES * ROWS_PER_WAVE];

  const int tid  = threadIdx.x;
  const int wave = tid >> 5;
  const int lane = tid & 31;
  // each wave owns a contiguous 64-row span, split into 4 tiles of 16 rows
  const int waveRowBase = blockIdx.x * ROWS_PER_BLOCK
                        + wave * (ROWS_PER_WAVE * TILES);

  // ---- stage weights (f32 -> f16) and LN params into LDS, once per block ----
  for (int i = tid; i < TDc * TZc; i += blockDim.x) sWz[i] = (_Float16)Wz[i];
  for (int i = tid; i < TZc * TDc; i += blockDim.x) sWu[i] = (_Float16)Wu[i];
  for (int i = tid; i < TZc; i += blockDim.x) { sZg[i] = zg[i]; sZb[i] = zb[i]; }
  for (int i = tid; i < TDc; i += blockDim.x) { sVg[i] = vg[i]; sVb[i] = vb[i]; }
  __syncthreads();

  // prefetch tile 0 (global_prefetch_b8)
  #pragma unroll
  for (int m = 0; m < ROWS_PER_WAVE; ++m)
    __builtin_prefetch(z + (size_t)(waveRowBase + m) * TZc + lane * 4, 0, 1);

  _Float16* myZn = sZn + wave * ROWS_PER_WAVE * TZc;
  _Float16* myRv = sRv + wave * ROWS_PER_WAVE * TDc;

  for (int tile = 0; tile < TILES; ++tile) {
    const int rowbase = waveRowBase + tile * ROWS_PER_WAVE;
    const float* zrow = z + (size_t)rowbase * TZc;

    // prefetch next tile while this one computes
    if (tile + 1 < TILES) {
      #pragma unroll
      for (int m = 0; m < ROWS_PER_WAVE; ++m)
        __builtin_prefetch(zrow + (size_t)(ROWS_PER_WAVE + m) * TZc + lane * 4, 0, 1);
    }

    // ---- distogram bucket per row (lane m in 0..15 owns row m) ----
    {
      const int m = lane & 15;
      const int r = rowbase + m;
      const int i = r >> 10, j = r & (NTOK - 1);
      const float dx = coords[3 * i + 0] - coords[3 * j + 0];
      const float dy = coords[3 * i + 1] - coords[3 * j + 1];
      const float dz = coords[3 * i + 2] - coords[3 * j + 2];
      const float d  = sqrtf(dx * dx + dy * dy + dz * dz);
      const float step = (50.75f - 3.25f) / 36.0f;   // 37 boundaries
      int idx = 0;
      #pragma unroll
      for (int k = 0; k < NBINS - 1; ++k) idx += (d > (3.25f + step * (float)k)) ? 1 : 0;
      if (lane < 16) { sIdx[wave * 16 + m] = idx; sMsk[wave * 16 + m] = tdm[r]; }
      // DS ops are in-order within a wave: stores visible to the reads below
    }

    // ---- LayerNorm over TZ=128, stage f16 rows (512B NT streaming loads) ----
    for (int m = 0; m < ROWS_PER_WAVE; ++m) {
      const v4f x = __builtin_nontemporal_load(
          (const v4f*)(zrow + (size_t)m * TZc) + lane);
      float s = x[0] + x[1] + x[2] + x[3];
      float q = x[0] * x[0] + x[1] * x[1] + x[2] * x[2] + x[3] * x[3];
      #pragma unroll
      for (int mk = 16; mk >= 1; mk >>= 1) {
        s += __shfl_xor(s, mk, 32);
        q += __shfl_xor(q, mk, 32);
      }
      const float mean = s * (1.0f / TZc);
      const float var  = q * (1.0f / TZc) - mean * mean;
      const float inv  = rsqrtf(var + 1e-5f);
      const int c0 = lane * 4;
      _Float16* dst = myZn + m * TZc + c0;
      #pragma unroll
      for (int e = 0; e < 4; ++e)
        dst[e] = (_Float16)((x[e] - mean) * inv * sZg[c0 + e] + sZb[c0 + e]);
    }

    // ---- GEMM1: v(16x64) = znorm(16x128) x Wz^T(128x64), C preloaded a_ij ----
    // C layout: VGPR t, lanes 0-15 -> (M=t, N=lane), lanes 16-31 -> (M=t+8, N=lane-16)
    v8f acc[4];
    #pragma unroll
    for (int n = 0; n < 4; ++n) {
      #pragma unroll
      for (int t = 0; t < 8; ++t) {
        const int M = t + ((lane & 16) ? 8 : 0);
        const int d = n * 16 + (lane & 15);
        acc[n][t] = sMsk[wave * 16 + M] * Wa[d * NBINS + sIdx[wave * 16 + M]];
      }
    }
    #pragma unroll
    for (int kc = 0; kc < 4; ++kc) {
      const int k0 = kc * 32;
      const v16h a = lds_frag(myZn, TZc, k0, lane);
      #pragma unroll
      for (int n = 0; n < 4; ++n) {
        const v16h b = lds_frag(sWz + n * 16 * TZc, TZc, k0, lane);
        acc[n] = __builtin_amdgcn_wmma_f32_16x16x32_f16(
            false, a, false, b, (short)0, acc[n], false, false);
      }
    }

    // ---- v = 2v; LayerNorm over TD=64 in-fragment; relu; restage f16 ----
    // Each row's 64 values live in one 16-lane group: xor 1/2/4/8 reduces per row.
    #pragma unroll
    for (int t = 0; t < 8; ++t) {
      float s = 0.0f, q = 0.0f;
      #pragma unroll
      for (int n = 0; n < 4; ++n) {
        const float xv = 2.0f * acc[n][t];
        s += xv; q += xv * xv;
      }
      #pragma unroll
      for (int mk = 8; mk >= 1; mk >>= 1) {
        s += __shfl_xor(s, mk, 32);
        q += __shfl_xor(q, mk, 32);
      }
      const float mean = s * (1.0f / TDc);
      const float var  = q * (1.0f / TDc) - mean * mean;
      const float inv  = rsqrtf(var + 1e-5f);
      const int M = t + ((lane & 16) ? 8 : 0);
      #pragma unroll
      for (int n = 0; n < 4; ++n) {
        const int d = n * 16 + (lane & 15);
        const float xv = 2.0f * acc[n][t];
        const float vn = (xv - mean) * inv * sVg[d] + sVb[d];
        myRv[M * TDc + d] = (_Float16)fmaxf(vn, 0.0f);
      }
    }

    // ---- GEMM2: u(16x128) = relu_v(16x64) x Wu^T(64x128) ----
    v8f acc2[8];
    #pragma unroll
    for (int n = 0; n < 8; ++n) acc2[n] = v8f{0, 0, 0, 0, 0, 0, 0, 0};
    #pragma unroll
    for (int kc = 0; kc < 2; ++kc) {
      const int k0 = kc * 32;
      const v16h a = lds_frag(myRv, TDc, k0, lane);
      #pragma unroll
      for (int n = 0; n < 8; ++n) {
        const v16h b = lds_frag(sWu + n * 16 * TDc, TDc, k0, lane);
        acc2[n] = __builtin_amdgcn_wmma_f32_16x16x32_f16(
            false, a, false, b, (short)0, acc2[n], false, false);
      }
    }

    // ---- store u (f32, write-once -> non-temporal); 64B/halfwave segments ----
    #pragma unroll
    for (int t = 0; t < 8; ++t) {
      const int M = t + ((lane & 16) ? 8 : 0);
      float* orow = out + (size_t)(rowbase + M) * TZc + (lane & 15);
      #pragma unroll
      for (int n = 0; n < 8; ++n)
        __builtin_nontemporal_store(acc2[n][t], orow + n * 16);
    }
  }
}

extern "C" void kernel_launch(void* const* d_in, const int* in_sizes, int n_in,
                              void* d_out, int out_size, void* d_ws, size_t ws_size,
                              hipStream_t stream) {
  const float* z   = (const float*)d_in[0];   // [1,1024,1024,128]
  const float* cc  = (const float*)d_in[1];   // [1,1024,3]
  const float* tdm = (const float*)d_in[2];   // [1,1024,1024]
  // d_in[3] = pair_mask (unused by reference)
  const float* Wz  = (const float*)d_in[4];   // [64,128]
  const float* Wa  = (const float*)d_in[5];   // [64,38]
  const float* Wu  = (const float*)d_in[6];   // [128,64]
  const float* zg  = (const float*)d_in[7];
  const float* zb  = (const float*)d_in[8];
  const float* vg  = (const float*)d_in[9];
  const float* vb  = (const float*)d_in[10];
  float* out = (float*)d_out;

  const int nPairs = NTOK * NTOK;
  dim3 grid(nPairs / ROWS_PER_BLOCK);   // 1M / 256 = 4096 blocks
  dim3 block(32 * WAVES);
  token_distance_fused<<<grid, block, 0, stream>>>(z, cc, tdm, Wz, Wa, Wu,
                                                   zg, zb, vg, vb, out);
}
